// MPNNModel_12249246728504
// MI455X (gfx1250) — matrix operand order repaired and checked by reading, not compiled, not run on hardware
//
#include <hip/hip_runtime.h>
#include <hip/hip_bf16.h>
#include <math.h>

// ---------------------------------------------------------------------------
// Types for CDNA5 WMMA (wave32, 16x16x32 bf16 -> f32)
// ---------------------------------------------------------------------------
typedef __bf16 bf16_t;
typedef bf16_t v16bf __attribute__((ext_vector_type(16)));
typedef bf16_t v8bf  __attribute__((ext_vector_type(8)));
typedef float  v8f   __attribute__((ext_vector_type(8)));

union Frag {
  v16bf v;
  v8bf  h[2];
  unsigned u[8];
};

__device__ __forceinline__ v8f wmma_bf16(v16bf a, v16bf b, v8f c) {
  // 8 args: (neg_a, A, neg_b, B, c_mod, C, reuse_a, reuse_b)
  return __builtin_amdgcn_wmma_f32_16x16x32_bf16(false, a, false, b, (short)0, c,
                                                 false, false);
}

// v_pk_mul_bf16: packed bf16 multiply (VOP3P op 42) — scale A fragment by h[e,k]
__device__ __forceinline__ unsigned pk_mul_bf16(unsigned a, unsigned b) {
  unsigned d;
  asm volatile("v_pk_mul_bf16 %0, %1, %2" : "=v"(d) : "v"(a), "v"(b));
  return d;
}

__device__ __forceinline__ unsigned bf16_splat_bits(bf16_t x) {
  union { bf16_t b; unsigned short s; } u;
  u.b = x;
  return ((unsigned)u.s << 16) | (unsigned)u.s;
}

// Async copy 16B global -> LDS (gfx1250 GLOBAL_LOAD_ASYNC_TO_LDS_B128).
// lds_off: per-lane LDS byte address (low 32 bits of a generic LDS pointer,
// per the aperture mapping). g: 64-bit global VA. Tracked by ASYNCcnt.
__device__ __forceinline__ void async_ld_b128(unsigned lds_off, const void* g) {
  asm volatile("global_load_async_to_lds_b128 %0, %1, off"
               :: "v"(lds_off), "v"(g)
               : "memory");
}
__device__ __forceinline__ void async_wait0() {
  asm volatile("s_wait_asynccnt 0x0" ::: "memory");
}
__device__ __forceinline__ unsigned lds_addr32(const void* p) {
  return (unsigned)(size_t)p;  // generic LDS pointer: addr[31:0] == LDS offset
}

// A fragment (16x32 MxK bf16) from row-major tile: lane<16 -> M=lane, K runs at
// hi*8 and 16+hi*8 (two contiguous 16B runs).
__device__ __forceinline__ Frag load_a_frag(const bf16_t* base, int lane,
                                            int stride, int kbase) {
  int m  = lane & 15;
  int hi = (lane >> 4) & 1;
  const bf16_t* p = base + m * stride + kbase + hi * 8;
  Frag f;
  f.h[0] = *(const v8bf*)(p);
  f.h[1] = *(const v8bf*)(p + 16);
  return f;
}

// B fragment (32x16 KxN bf16) from transposed [N][K] storage: lane holds
// column N = nbase + lane%16, K runs at hi*8 and 16+hi*8 (two 16B runs).
__device__ __forceinline__ Frag load_b_frag(const bf16_t* bT, int lane,
                                            int rowStride, int nbase,
                                            long kbase) {
  int n  = nbase + (lane & 15);
  int hi = (lane >> 4) & 1;
  const bf16_t* p = bT + (long)n * rowStride + kbase + hi * 8;
  Frag f;
  f.h[0] = *(const v8bf*)(p);
  f.h[1] = *(const v8bf*)(p + 16);
  return f;
}

// ---------------------------------------------------------------------------
// Weight prep
// ---------------------------------------------------------------------------
// in: [R][64] f32 row-major -> out: [64][R] bf16 (transpose + convert)
__global__ __launch_bounds__(256) void transpose_to_bf16(
    const float* __restrict__ in, bf16_t* __restrict__ out, int R) {
  int idx = blockIdx.x * blockDim.x + threadIdx.x;
  if (idx >= R * 64) return;
  int r = idx >> 6;
  int o = idx & 63;
  out[(long)o * R + r] = (bf16_t)in[idx];
}

// eW1 [16][64] f32 -> [64][32] bf16, K padded 16->32 with zeros
__global__ __launch_bounds__(256) void conv_w1(const float* __restrict__ w1,
                                               bf16_t* __restrict__ out) {
  int idx = blockIdx.x * blockDim.x + threadIdx.x;
  if (idx >= 64 * 32) return;
  int o = idx >> 5;
  int k = idx & 31;
  out[idx] = (k < 16) ? (bf16_t)w1[k * 64 + o] : (bf16_t)0.0f;
}

__global__ __launch_bounds__(256) void zero_f32(float* __restrict__ p, int n) {
  int i = blockIdx.x * blockDim.x + threadIdx.x;
  if (i < n) p[i] = 0.0f;
}

// ---------------------------------------------------------------------------
// Edge MLP stage 1: h = relu(edge_attr @ eW1 + eb1)   -> bf16 [E][64]
// ---------------------------------------------------------------------------
__global__ __launch_bounds__(128) void edge_h_kernel(
    const float* __restrict__ eattr,  // [E,16]
    const bf16_t* __restrict__ W1bT,  // [64][32]
    const float* __restrict__ eb1,    // [64]
    bf16_t* __restrict__ hout,        // [E,64]
    int nTiles) {
  __shared__ bf16_t at[4][16 * 32];
  int lane = threadIdx.x & 31, wid = threadIdx.x >> 5;
  int tile = blockIdx.x * 4 + wid;
  bool active = tile < nTiles;
  int e0 = tile * 16;
  if (active) {
    for (int idx = lane; idx < 16 * 32; idx += 32) {
      int m = idx >> 5, kk = idx & 31;
      at[wid][idx] = (kk < 16) ? (bf16_t)eattr[(long)(e0 + m) * 16 + kk]
                               : (bf16_t)0.0f;
    }
  }
  __syncthreads();
  if (!active) return;

  Frag a = load_a_frag(&at[wid][0], lane, 32, 0);
  int n = lane & 15, hi = (lane >> 4) & 1;
#pragma unroll
  for (int nt = 0; nt < 4; ++nt) {
    Frag b = load_b_frag(W1bT, lane, 32, nt * 16, 0);
    v8f acc;
#pragma unroll
    for (int q = 0; q < 8; ++q) acc[q] = 0.0f;
    acc = wmma_bf16(a.v, b.v, acc);
    float bias = eb1[nt * 16 + n];
#pragma unroll
    for (int j = 0; j < 8; ++j) {
      float v2 = fmaxf(acc[j] + bias, 0.0f);
      hout[(long)(e0 + j + hi * 8) * 64 + nt * 16 + n] = (bf16_t)v2;
    }
  }
}

// ---------------------------------------------------------------------------
// Fused message kernel:
//   msg[e,o] = sum_{k,i} h[e,k]*x[src[e],i]*W2[k,i,o] + sum_i x_src[i]*B2[i,o]
// = Z @ W2'  with Z built on the fly (A frag = h[e,k] * x-frag via
// v_pk_mul_bf16). Per block: 4 waves x 16-edge tiles share a double-buffered
// LDS stage of the IN-wide W2' K-chunk, filled with async global->LDS copies.
// ---------------------------------------------------------------------------
template <int IN>
__global__ __launch_bounds__(128) void msg_kernel(
    const float* __restrict__ xin,    // [N, IN]
    const int* __restrict__ src,      // [E]
    const int* __restrict__ dst,      // [E]
    const bf16_t* __restrict__ hbuf,  // [E][64] bf16
    const bf16_t* __restrict__ W2bT,  // [64][64*IN] bf16 (transposed)
    const bf16_t* __restrict__ B2T,   // [64][IN] bf16 (transposed eb2)
    float* __restrict__ sums,         // [N,64]
    float* __restrict__ cnt,          // [N]
    int nTiles) {
  const int KF = 64 * IN;
  const int NC = IN / 32;
  __shared__ bf16_t xt[4][16 * IN];
  __shared__ bf16_t ht[4][16 * 64];
  __shared__ bf16_t bbuf[2][NC * 64 * 32];  // double-buffered B chunk [c][o][32]
  int lane = threadIdx.x & 31, wid = threadIdx.x >> 5;
  int tile = blockIdx.x * 4 + wid;
  bool active = tile < nTiles;
  int e0 = tile * 16;

  // Stage per-wave x tile (gather + f32->bf16 convert) and h tile (async copy).
  if (active) {
    for (int idx = lane; idx < 16 * IN; idx += 32) {
      int m = idx / IN, i = idx - m * IN;
      int s = src[e0 + m];
      xt[wid][idx] = (bf16_t)xin[(long)s * IN + i];
    }
    const bf16_t* hp = hbuf + (long)e0 * 64;
    unsigned hl = lds_addr32(&ht[wid][0]);
    for (int q = lane; q < 128; q += 32)  // 128 x 16B = 2KB
      async_ld_b128(hl + (unsigned)q * 16u, hp + q * 8);
  }

  // Block-cooperative async stage of W2' chunk for hidden unit k into bbuf[buf]
  auto stage_b = [&](int k, int buf) {
    unsigned lbase = lds_addr32(&bbuf[buf][0]);
    for (int q = (int)threadIdx.x; q < NC * 256; q += 128) {
      int c = q >> 8;       // K-subchunk (i-range)
      int r = q & 255;
      int o = r >> 2;       // output column / B^T row
      int part = r & 3;     // 16B slice within the 64B row chunk
      const bf16_t* gp = W2bT + (long)o * KF + (long)k * IN + c * 32 + part * 8;
      unsigned loff = lbase + 2u * (unsigned)((c * 64 + o) * 32 + part * 8);
      async_ld_b128(loff, gp);
    }
  };

  stage_b(0, 0);

  // Wait for x-tile stores + first async batch, then build reusable fragments.
  async_wait0();
  __syncthreads();

  Frag xf[NC];
#pragma unroll
  for (int c = 0; c < NC; ++c)
    xf[c] = load_a_frag(&xt[wid][0], lane, IN, c * 32);
  const bf16_t* hrow = &ht[wid][(lane & 15) * 64];

  v8f acc[4];
#pragma unroll
  for (int nt = 0; nt < 4; ++nt)
#pragma unroll
    for (int q = 0; q < 8; ++q) acc[nt][q] = 0.0f;

  // Main GEMM: K_flat = 64*IN, chunked by IN (one hidden unit k per chunk).
  int buf = 0;
  for (int k = 0; k < 64; ++k) {
    if (k > 0) {
      async_wait0();     // own async loads for this chunk complete
      __syncthreads();   // -> whole chunk visible block-wide
    }
    if (k + 1 < 64) stage_b(k + 1, buf ^ 1);  // prefetch next chunk

    unsigned hss = bf16_splat_bits(hrow[k]);
    const bf16_t* bb = &bbuf[buf][0];
#pragma unroll
    for (int c = 0; c < NC; ++c) {
      Frag a;
#pragma unroll
      for (int q = 0; q < 8; ++q) a.u[q] = pk_mul_bf16(xf[c].u[q], hss);
#pragma unroll
      for (int nt = 0; nt < 4; ++nt) {
        Frag b = load_b_frag(bb + c * 64 * 32, lane, 32, nt * 16, 0);
        acc[nt] = wmma_bf16(a.v, b.v, acc[nt]);
      }
    }
    buf ^= 1;
  }
  // eb2 bias term: x_src @ B2 (small, straight from global)
#pragma unroll
  for (int c = 0; c < NC; ++c)
#pragma unroll
    for (int nt = 0; nt < 4; ++nt) {
      Frag b = load_b_frag(B2T, lane, IN, nt * 16, c * 32);
      acc[nt] = wmma_bf16(xf[c].v, b.v, acc[nt]);
    }

  // Scatter: element j of acc[nt] -> edge e0 + j + 8*hi, column nt*16 + n
  if (active) {
    int n = lane & 15, hi = (lane >> 4) & 1;
#pragma unroll
    for (int j = 0; j < 8; ++j) {
      int e = e0 + j + hi * 8;
      int d = dst[e];
      float* srow = sums + (long)d * 64 + n;
#pragma unroll
      for (int nt = 0; nt < 4; ++nt) atomicAdd(srow + nt * 16, acc[nt][j]);
      if (n == 0) atomicAdd(&cnt[d], 1.0f);
    }
  }
}

// ---------------------------------------------------------------------------
// Node update: xout = relu( sums/max(cnt,1) + xin @ root + bias )
// ---------------------------------------------------------------------------
template <int IN>
__global__ __launch_bounds__(128) void node_update_kernel(
    const float* __restrict__ xin,     // [N, IN]
    const bf16_t* __restrict__ rootT,  // [64][IN]
    const float* __restrict__ bias,    // [64]
    const float* __restrict__ sums,    // [N,64]
    const float* __restrict__ cnt,     // [N]
    float* __restrict__ xout,          // [N,64]
    int nTiles) {
  const int NC = IN / 32;
  __shared__ bf16_t xt[4][16 * IN];
  int lane = threadIdx.x & 31, wid = threadIdx.x >> 5;
  int tile = blockIdx.x * 4 + wid;
  bool active = tile < nTiles;
  int v0 = tile * 16;
  if (active) {
    for (int idx = lane; idx < 16 * IN; idx += 32) {
      int m = idx / IN, i = idx - m * IN;
      xt[wid][idx] = (bf16_t)xin[(long)(v0 + m) * IN + i];
    }
  }
  __syncthreads();
  if (!active) return;

  Frag xf[NC];
#pragma unroll
  for (int c = 0; c < NC; ++c)
    xf[c] = load_a_frag(&xt[wid][0], lane, IN, c * 32);

  v8f acc[4];
#pragma unroll
  for (int nt = 0; nt < 4; ++nt)
#pragma unroll
    for (int q = 0; q < 8; ++q) acc[nt][q] = 0.0f;

#pragma unroll
  for (int c = 0; c < NC; ++c)
#pragma unroll
    for (int nt = 0; nt < 4; ++nt) {
      Frag b = load_b_frag(rootT, lane, IN, nt * 16, c * 32);
      acc[nt] = wmma_bf16(xf[c].v, b.v, acc[nt]);
    }

  int n = lane & 15, hi = (lane >> 4) & 1;
#pragma unroll
  for (int j = 0; j < 8; ++j) {
    int v = v0 + j + hi * 8;
    float cn = fmaxf(cnt[v], 1.0f);
#pragma unroll
    for (int nt = 0; nt < 4; ++nt) {
      int o = nt * 16 + n;
      float r = acc[nt][j] + sums[(long)v * 64 + o] / cn + bias[o];
      xout[(long)v * 64 + o] = fmaxf(r, 0.0f);
    }
  }
}

// ---------------------------------------------------------------------------
// Graph readout: segment mean/max over batch (post-relu values >= 0, so
// int-pattern atomicMax is order-preserving; 0-init matches the reference's
// isfinite clamp for empty segments).
// ---------------------------------------------------------------------------
__global__ __launch_bounds__(256) void readout_kernel(
    const float* __restrict__ x,    // [N,64]
    const int* __restrict__ batch,  // [N]
    float* __restrict__ gsum,       // [G,64]
    int* __restrict__ gmax,         // [G,64] (float bits)
    float* __restrict__ gcnt,       // [G]
    int total) {
  int idx = blockIdx.x * blockDim.x + threadIdx.x;
  if (idx >= total) return;
  int v = idx >> 6, ch = idx & 63;
  int b = batch[v];
  float val = x[idx];
  atomicAdd(&gsum[b * 64 + ch], val);
  atomicMax(&gmax[b * 64 + ch], __float_as_int(val));
  if (ch == 0) atomicAdd(&gcnt[b], 1.0f);
}

__global__ __launch_bounds__(64) void classifier_kernel(
    const float* __restrict__ gsum, const int* __restrict__ gmax,
    const float* __restrict__ gcnt, const float* __restrict__ cW1,
    const float* __restrict__ cb1, const float* __restrict__ cW2,
    const float* __restrict__ cb2, float* __restrict__ out) {
  int g = threadIdx.x;
  if (g >= 64) return;
  float cn = fmaxf(gcnt[g], 1.0f);
  float gv[128];
  for (int t = 0; t < 64; ++t) gv[t] = gsum[g * 64 + t] / cn;
  for (int t = 0; t < 64; ++t) gv[64 + t] = __int_as_float(gmax[g * 64 + t]);
  float logit = cb2[0];
  for (int j = 0; j < 64; ++j) {
    float hh = cb1[j];
    for (int t = 0; t < 128; ++t) hh += gv[t] * cW1[t * 64 + j];
    hh = fmaxf(hh, 0.0f);
    logit += hh * cW2[j];
  }
  out[g] = 1.0f / (1.0f + expf(-logit));
}

// ---------------------------------------------------------------------------
// Launch
// ---------------------------------------------------------------------------
extern "C" void kernel_launch(void* const* d_in, const int* in_sizes, int n_in,
                              void* d_out, int out_size, void* d_ws,
                              size_t ws_size, hipStream_t stream) {
  const float* x      = (const float*)d_in[0];
  const int*   ei     = (const int*)d_in[1];
  const float* eattr  = (const float*)d_in[2];
  const int*   batch  = (const int*)d_in[3];
  const float* eW1_0  = (const float*)d_in[4];
  const float* eb1_0  = (const float*)d_in[5];
  const float* eW2_0  = (const float*)d_in[6];
  const float* eb2_0  = (const float*)d_in[7];
  const float* root_0 = (const float*)d_in[8];
  const float* bias_0 = (const float*)d_in[9];
  const float* eW1_1  = (const float*)d_in[10];
  const float* eb1_1  = (const float*)d_in[11];
  const float* eW2_1  = (const float*)d_in[12];
  const float* eb2_1  = (const float*)d_in[13];
  const float* root_1 = (const float*)d_in[14];
  const float* bias_1 = (const float*)d_in[15];
  const float* cW1    = (const float*)d_in[16];
  const float* cb1    = (const float*)d_in[17];
  const float* cW2    = (const float*)d_in[18];
  const float* cb2    = (const float*)d_in[19];

  const int E  = in_sizes[2] / 16;  // edge_attr [E,16]
  const int NN = in_sizes[0] / 32;  // x [N,32]
  const int G  = 64;
  const int* src = ei;
  const int* dst = ei + E;

  char* w = (char*)d_ws;
  size_t off = 0;
  auto alloc = [&](size_t bytes) -> char* {
    char* p = w + off;
    off = (off + bytes + 255) & ~(size_t)255;
    return p;
  };

  bf16_t* hbuf   = (bf16_t*)alloc((size_t)E * 64 * 2);
  bf16_t* W1bT0  = (bf16_t*)alloc(64 * 32 * 2);
  bf16_t* W1bT1  = (bf16_t*)alloc(64 * 32 * 2);
  bf16_t* W2bT0  = (bf16_t*)alloc((size_t)64 * 2048 * 2);
  bf16_t* W2bT1  = (bf16_t*)alloc((size_t)64 * 4096 * 2);
  bf16_t* B2T0   = (bf16_t*)alloc(64 * 32 * 2);
  bf16_t* B2T1   = (bf16_t*)alloc(64 * 64 * 2);
  bf16_t* rootT0 = (bf16_t*)alloc(64 * 32 * 2);
  bf16_t* rootT1 = (bf16_t*)alloc(64 * 64 * 2);
  float* sums = (float*)alloc((size_t)NN * 64 * 4 + (size_t)NN * 4);
  float* cnt  = sums + (size_t)NN * 64;
  float* x1   = (float*)alloc((size_t)NN * 64 * 4);
  float* x2   = (float*)alloc((size_t)NN * 64 * 4);
  float* gsum = (float*)alloc((size_t)G * 64 * 4 * 2 + G * 4);
  int*   gmax = (int*)(gsum + G * 64);
  float* gcnt = (float*)(gsum + 2 * G * 64);

  // --- weight prep ---
  transpose_to_bf16<<<(2048 * 64 + 255) / 256, 256, 0, stream>>>(eW2_0, W2bT0, 2048);
  transpose_to_bf16<<<(4096 * 64 + 255) / 256, 256, 0, stream>>>(eW2_1, W2bT1, 4096);
  transpose_to_bf16<<<(32 * 64 + 255) / 256, 256, 0, stream>>>(eb2_0, B2T0, 32);
  transpose_to_bf16<<<(64 * 64 + 255) / 256, 256, 0, stream>>>(eb2_1, B2T1, 64);
  transpose_to_bf16<<<(32 * 64 + 255) / 256, 256, 0, stream>>>(root_0, rootT0, 32);
  transpose_to_bf16<<<(64 * 64 + 255) / 256, 256, 0, stream>>>(root_1, rootT1, 64);
  conv_w1<<<(64 * 32 + 255) / 256, 256, 0, stream>>>(eW1_0, W1bT0);
  conv_w1<<<(64 * 32 + 255) / 256, 256, 0, stream>>>(eW1_1, W1bT1);

  const int eTiles = E / 16;
  const int nTiles = NN / 16;
  const int eBlocks = (eTiles + 3) / 4;
  const int nBlocks = (nTiles + 3) / 4;
  const int zN = NN * 64 + NN;

  // --- layer 0 ---
  zero_f32<<<(zN + 255) / 256, 256, 0, stream>>>(sums, zN);
  edge_h_kernel<<<eBlocks, 128, 0, stream>>>(eattr, W1bT0, eb1_0, hbuf, eTiles);
  msg_kernel<32><<<eBlocks, 128, 0, stream>>>(x, src, dst, hbuf, W2bT0, B2T0,
                                              sums, cnt, eTiles);
  node_update_kernel<32><<<nBlocks, 128, 0, stream>>>(x, rootT0, bias_0, sums,
                                                      cnt, x1, nTiles);
  // --- layer 1 ---
  zero_f32<<<(zN + 255) / 256, 256, 0, stream>>>(sums, zN);
  edge_h_kernel<<<eBlocks, 128, 0, stream>>>(eattr, W1bT1, eb1_1, hbuf, eTiles);
  msg_kernel<64><<<eBlocks, 128, 0, stream>>>(x1, src, dst, hbuf, W2bT1, B2T1,
                                              sums, cnt, eTiles);
  node_update_kernel<64><<<nBlocks, 128, 0, stream>>>(x1, rootT1, bias_1, sums,
                                                      cnt, x2, nTiles);
  // --- readout + classifier ---
  const int gZ = G * 64 * 2 + G;
  zero_f32<<<(gZ + 255) / 256, 256, 0, stream>>>(gsum, gZ);
  readout_kernel<<<(NN * 64 + 255) / 256, 256, 0, stream>>>(x2, batch, gsum,
                                                            gmax, gcnt, NN * 64);
  classifier_kernel<<<1, 64, 0, stream>>>(gsum, gmax, gcnt, cW1, cb1, cW2, cb2,
                                          (float*)d_out);
}
